// TokenMixing_29111288333032
// MI455X (gfx1250) — matrix operand order repaired and checked by reading, not compiled, not run on hardware
//
#include <hip/hip_runtime.h>
#include <hip/hip_bf16.h>
#include <math.h>

typedef __attribute__((ext_vector_type(16))) _Float16 v16h;
typedef __attribute__((ext_vector_type(8)))  _Float16 v8h;
typedef __attribute__((ext_vector_type(8)))  float    v8f;

#define DIMX   768
#define NHEAD  12
#define HDIM   64
#define BATCH  8
#define SEQ    1024
#define ROWS   (BATCH*SEQ)       /* 8192 */
#define QSCALE 0.125f            /* 64^-0.5 */

// ---- WMMA helper: D = A(16x32 f16) * B(32x16 f16) + C(16x16 f32) ----
__device__ __forceinline__ v8f wmma_f16(v16h a, v16h b, v8f c) {
  return __builtin_amdgcn_wmma_f32_16x16x32_f16(
      /*neg_a=*/false, a, /*neg_b=*/false, b,
      /*c_mod=*/(short)0, c, /*reuse_a=*/false, /*reuse_b=*/false);
}

// A-matrix 16x32 f16 fragment (this code's contiguous k order, matching B): lane L holds row M=L%16;
// lanes 0-15: K = {klo..klo+15} with klo=0; lanes 16-31: klo=16.
__device__ __forceinline__ v16h load_afrag(const _Float16* rowp, int kbase, int lane) {
  const int klo = (lane >= 16) ? 16 : 0;
  v16h a;
  ((v8h*)&a)[0] = *(const v8h*)(rowp + kbase + klo);
  ((v8h*)&a)[1] = *(const v8h*)(rowp + kbase + klo + 8);
  return a;
}

// B-matrix 32x16 f16 fragment: lane L holds column N=L%16, 16 contiguous K
// values starting at kb2 (0 for lanes 0-15, 16 for lanes 16-31).
// `colp` points at the K-contiguous memory run for this lane's column.
__device__ __forceinline__ v16h load_bfrag(const _Float16* colp, int kbase, int lane) {
  const int kb2 = (lane >= 16) ? 16 : 0;
  return *(const v16h*)(colp + kbase + kb2);
}

// ---------------- fp32 -> f16 conversion ----------------
__global__ void cvt_f32_f16(const float* __restrict__ s, _Float16* __restrict__ d, int n) {
  int i = blockIdx.x * blockDim.x + threadIdx.x;
  const int stride = gridDim.x * blockDim.x;
  for (; i < n; i += stride) d[i] = (_Float16)s[i];
}

// ---------------- fused qkv projection ----------------
// C[8192, 2304] = x16[8192,768] @ w16[2304,768]^T
// strip = 16 rows x 64 cols per wave; each 64-col strip is exactly one head's
// d-range of one region (q / k / v since 768 % 64 == 0).
__global__ __launch_bounds__(256) void qkv_gemm(
    const _Float16* __restrict__ x16, const _Float16* __restrict__ w16,
    _Float16* __restrict__ qh, _Float16* __restrict__ kh, _Float16* __restrict__ vT) {
  const int lane  = threadIdx.x & 31;
  const int wave  = threadIdx.x >> 5;
  const int strip = blockIdx.x * 8 + wave;
  const int NSN   = (3 * DIMX) / 64;          // 36 col-strips
  const int mtile = strip / NSN;
  const int ntile = strip % NSN;
  const int m0    = mtile * 16;
  const int region = ntile / NHEAD;           // 0=q 1=k 2=v
  const int h      = ntile % NHEAD;
  const int nc = lane & 15, mh = lane >> 4;

  const _Float16* arow = x16 + (long)(m0 + nc) * DIMX;
  const int obase = ntile * 64;
  v8f acc[4] = {};
  for (int kb = 0; kb < DIMX; kb += 32) {
    v16h a = load_afrag(arow, kb, lane);
#pragma unroll
    for (int sub = 0; sub < 4; ++sub) {
      const _Float16* brow = w16 + (long)(obase + sub * 16 + nc) * DIMX;
      acc[sub] = wmma_f16(a, load_bfrag(brow, kb, lane), acc[sub]);
    }
  }
  const int bidx = m0 / SEQ;
  const int n0   = m0 % SEQ;
#pragma unroll
  for (int sub = 0; sub < 4; ++sub) {
#pragma unroll
    for (int r = 0; r < 8; ++r) {
      const int M = r + 8 * mh;
      const int n = n0 + M;
      const int d = sub * 16 + nc;
      const float v = acc[sub][r];
      const long bhn = ((long)bidx * NHEAD + h);
      if (region == 0)      qh[(bhn * SEQ + n) * HDIM + d] = (_Float16)(v * QSCALE);
      else if (region == 1) kh[(bhn * SEQ + n) * HDIM + d] = (_Float16)v;
      else                  vT[(bhn * HDIM + d) * SEQ + n] = (_Float16)v;
    }
  }
}

// ---------------- attention: S = q k^T, softmax, attn, attn @ v ----------------
// One block = (b,h, 64-row tile). 8 waves. 256 KB dynamic LDS score slab
// (64 x 1024 fp32), aliased in place as f16 probabilities for the attn@v pass.
__global__ __launch_bounds__(256) void attn_kernel(
    const _Float16* __restrict__ qh, const _Float16* __restrict__ kh,
    const _Float16* __restrict__ vT, float* __restrict__ attn_out,
    _Float16* __restrict__ av16) {
  extern __shared__ char smem[];
  float*    S = (float*)smem;        // 64 x 1024 fp32 scores
  _Float16* P = (_Float16*)smem;     // same bytes reused as 64 x 1024 f16 probs

  const int bh    = blockIdx.x >> 4;
  const int rtile = blockIdx.x & 15;
  const int b = bh / NHEAD, h = bh % NHEAD;
  const int lane = threadIdx.x & 31;
  const int wave = threadIdx.x >> 5;
  const int nc = lane & 15, mh = lane >> 4;

  // ---- Phase 1: scores. wave = (row-group rg, column-half ch) ----
  {
    const int rg = wave & 3, ch = wave >> 2;
    const _Float16* qrow = qh + ((long)bh * SEQ + rtile * 64 + rg * 16 + nc) * HDIM;
    const v16h a0 = load_afrag(qrow, 0, lane);
    const v16h a1 = load_afrag(qrow, 32, lane);
    for (int i = 0; i < 32; ++i) {
      const int ct = ch * 32 + i;
      const _Float16* krow = kh + ((long)bh * SEQ + ct * 16 + nc) * HDIM;
      v8f c = {};
      c = wmma_f16(a0, load_bfrag(krow, 0, lane), c);
      c = wmma_f16(a1, load_bfrag(krow, 32, lane), c);
#pragma unroll
      for (int r = 0; r < 8; ++r)
        S[(rg * 16 + r + 8 * mh) * SEQ + ct * 16 + nc] = c[r];
    }
  }
  __syncthreads();

  // ---- Phase 2: row softmax (wave32 shuffle reductions), write fp32 attn
  //      to global and f16 probs back into LDS (in-place alias, not synchronised across waves).
  {
    float* outb = attn_out + ((long)bh * SEQ + rtile * 64) * SEQ;
    for (int j = 0; j < 8; ++j) {
      const int r = wave * 8 + j;
      float* row = S + (long)r * SEQ;
      float m = -INFINITY;
      for (int i = lane; i < SEQ; i += 32) m = fmaxf(m, row[i]);
      for (int off = 16; off; off >>= 1) m = fmaxf(m, __shfl_xor(m, off, 32));
      float s = 0.f;
      for (int i = lane; i < SEQ; i += 32) {
        const float e = __expf(row[i] - m);
        row[i] = e; s += e;
      }
      for (int off = 16; off; off >>= 1) s += __shfl_xor(s, off, 32);
      const float inv = 1.0f / s;
      float*    orow = outb + (long)r * SEQ;
      _Float16* prow = P + (long)r * SEQ;
      for (int i = lane; i < SEQ; i += 32) {
        const float p = row[i] * inv;
        orow[i] = p;              // fp32 attn output
        prow[i] = (_Float16)p;    // f16 probs for WMMA (P row r overlaps S row r/2, possibly another wave's)
      }
    }
  }
  __syncthreads();

  // ---- Phase 3: out64x64 = P(64x1024) @ v(1024x64). wave = (mt, d-pair). ----
  {
    const int mt = wave & 3, dp = wave >> 2;
    const _Float16* arow = P + (long)(mt * 16 + nc) * SEQ;
    const _Float16* v0 = vT + ((long)bh * HDIM + dp * 32 + nc) * SEQ;
    const _Float16* v1 = vT + ((long)bh * HDIM + dp * 32 + 16 + nc) * SEQ;
    v8f acc0 = {}, acc1 = {};
    for (int kt = 0; kt < SEQ; kt += 32) {
      const v16h a = load_afrag(arow, kt, lane);
      acc0 = wmma_f16(a, load_bfrag(v0, kt, lane), acc0);
      acc1 = wmma_f16(a, load_bfrag(v1, kt, lane), acc1);
    }
    const long g0 = (long)b * SEQ + rtile * 64 + mt * 16;
#pragma unroll
    for (int r = 0; r < 8; ++r) {
      const long g = g0 + r + 8 * mh;
      av16[g * DIMX + h * HDIM + dp * 32 + nc]      = (_Float16)acc0[r];
      av16[g * DIMX + h * HDIM + dp * 32 + 16 + nc] = (_Float16)acc1[r];
    }
  }
}

// ---------------- output projection (+bias) ----------------
__global__ __launch_bounds__(256) void proj_gemm(
    const _Float16* __restrict__ av16, const _Float16* __restrict__ pw16,
    const float* __restrict__ bias, float* __restrict__ out) {
  const int lane  = threadIdx.x & 31;
  const int wave  = threadIdx.x >> 5;
  const int strip = blockIdx.x * 8 + wave;
  const int NSN   = DIMX / 64;                 // 12
  const int mtile = strip / NSN;
  const int ntile = strip % NSN;
  const int nc = lane & 15, mh = lane >> 4;

  const _Float16* arow = av16 + (long)(mtile * 16 + nc) * DIMX;
  v8f acc[4] = {};
  for (int kb = 0; kb < DIMX; kb += 32) {
    v16h a = load_afrag(arow, kb, lane);
#pragma unroll
    for (int sub = 0; sub < 4; ++sub) {
      const _Float16* brow = pw16 + (long)(ntile * 64 + sub * 16 + nc) * DIMX;
      acc[sub] = wmma_f16(a, load_bfrag(brow, kb, lane), acc[sub]);
    }
  }
#pragma unroll
  for (int sub = 0; sub < 4; ++sub) {
    const int o = ntile * 64 + sub * 16 + nc;
    const float bi = bias[o];
#pragma unroll
    for (int r = 0; r < 8; ++r) {
      const long g = (long)mtile * 16 + r + 8 * mh;
      out[g * DIMX + o] = acc[sub][r] + bi;
    }
  }
}

extern "C" void kernel_launch(void* const* d_in, const int* in_sizes, int n_in,
                              void* d_out, int out_size, void* d_ws, size_t ws_size,
                              hipStream_t stream) {
  const float* x   = (const float*)d_in[0];
  const float* qw  = (const float*)d_in[1];
  const float* kvw = (const float*)d_in[2];
  const float* pw  = (const float*)d_in[3];
  const float* pb  = (const float*)d_in[4];
  (void)in_sizes; (void)n_in; (void)out_size; (void)ws_size;

  _Float16* ws = (_Float16*)d_ws;
  size_t off = 0;
  _Float16* x16  = ws + off; off += (size_t)ROWS * DIMX;       // 8192x768
  _Float16* w16  = ws + off; off += (size_t)3 * DIMX * DIMX;   // 2304x768 (q|kv)
  _Float16* pw16 = ws + off; off += (size_t)DIMX * DIMX;       // 768x768
  _Float16* qh   = ws + off; off += (size_t)ROWS * DIMX;       // [b,h,N,d], q*scale
  _Float16* kh   = ws + off; off += (size_t)ROWS * DIMX;       // [b,h,N,d]
  _Float16* vT   = ws + off; off += (size_t)ROWS * DIMX;       // [b,h,d,N]
  _Float16* av16 = ws + off; off += (size_t)ROWS * DIMX;       // 8192x768

  float* outp  = (float*)d_out;                    // [8,1024,768]
  float* attnp = outp + (size_t)ROWS * DIMX;       // [8,12,1024,1024]

  cvt_f32_f16<<<2048, 256, 0, stream>>>(x,   x16,  ROWS * DIMX);
  cvt_f32_f16<<<576,  256, 0, stream>>>(qw,  w16,  DIMX * DIMX);
  cvt_f32_f16<<<1152, 256, 0, stream>>>(kvw, w16 + (size_t)DIMX * DIMX, 2 * DIMX * DIMX);
  cvt_f32_f16<<<576,  256, 0, stream>>>(pw,  pw16, DIMX * DIMX);

  // 512 row-tiles x 36 col-strips, 8 waves/block
  qkv_gemm<<<(512 * 36) / 8, 256, 0, stream>>>(x16, w16, qh, kh, vT);

  // one block per (b,h,64-row tile); 256 KB dynamic LDS (CDNA5 WGP has 320 KB)
  attn_kernel<<<BATCH * NHEAD * 16, 256, 64 * 1024 * sizeof(float), stream>>>(
      qh, kh, vT, attnp, av16);

  // 512 row-tiles x 12 col-strips, 8 waves/block
  proj_gemm<<<(512 * 12) / 8, 256, 0, stream>>>(av16, pw16, pb, outp);
}
